// MHA_25950192403155
// MI455X (gfx1250) — compile-verified, matrix operand order checked by reading, not run on hardware
//
#include <hip/hip_runtime.h>
#include <hip/hip_bf16.h>

// ---------------------------------------------------------------------------
// MHA forward for MI455X (gfx1250, wave32, WMMA + TDM).
// All matmuls in bf16 via v_wmma_f32_16x16x32_bf16 with f32 accumulation.
// GEMM stages block tiles into LDS with the Tensor Data Mover (double
// buffered: next tile's DMA in flight while current tile is in WMMA).
// ---------------------------------------------------------------------------

typedef __attribute__((ext_vector_type(16))) __bf16       v16bf;
typedef __attribute__((ext_vector_type(8)))  float        v8f;
typedef __attribute__((ext_vector_type(4)))  unsigned int u32x4;
typedef __attribute__((ext_vector_type(4)))  int          s32x4;
typedef __attribute__((ext_vector_type(8)))  int          s32x8;

#define B_BATCH 2
#define L_SEQ   4096
#define D_MODEL 512
#define N_HEADS 8
#define D_HEAD  64

#if defined(__has_builtin)
#if __has_builtin(__builtin_amdgcn_tensor_load_to_lds)
#define HAVE_TDM 1
#endif
#endif
#ifndef HAVE_TDM
#define HAVE_TDM 0
#endif

union FragBF16 {
    v16bf v;
    uint4 q[2];      // two 16-byte chunks per lane
};

__device__ __forceinline__ v8f wmma_bf16(const FragBF16& a, const FragBF16& b, v8f c) {
    return __builtin_amdgcn_wmma_f32_16x16x32_bf16(
        /*neg_a=*/false, a.v, /*neg_b=*/false, b.v,
        /*c_mod=*/(short)0, c, /*reuse_a=*/false, /*reuse_b=*/false);
}

// Reduction across a 16-lane half of the wave (xor masks 1,2,4,8 stay in-half).
__device__ __forceinline__ float halfgrp_max(float v) {
    v = fmaxf(v, __shfl_xor(v, 1, 32));
    v = fmaxf(v, __shfl_xor(v, 2, 32));
    v = fmaxf(v, __shfl_xor(v, 4, 32));
    v = fmaxf(v, __shfl_xor(v, 8, 32));
    return v;
}
__device__ __forceinline__ float halfgrp_sum(float v) {
    v += __shfl_xor(v, 1, 32);
    v += __shfl_xor(v, 2, 32);
    v += __shfl_xor(v, 4, 32);
    v += __shfl_xor(v, 8, 32);
    return v;
}

#if HAVE_TDM
// Issue a 2-D TDM tile load: rows x tile_elems (bf16) from a row-major tensor
// whose rows are row_stride_elems apart, into contiguous LDS at lds_off.
// D# layout per CDNA5 ISA ch.8: group0 = {count/flags, lds_addr, global_addr,
// type=2}; group1 = {data_size, tensor dims, tile dims, dim0 stride}.
__device__ __forceinline__ void tdm_load_tile_2d(const void* gaddr, unsigned lds_off,
                                                 unsigned rows, unsigned tile_elems,
                                                 unsigned row_stride_elems) {
    const unsigned long long ga = (unsigned long long)gaddr;
    u32x4 g0;
    g0[0] = 1u;                                             // count=1, user mode
    g0[1] = lds_off;                                        // lds_addr (bytes)
    g0[2] = (unsigned)(ga & 0xffffffffu);                   // global_addr[31:0]
    g0[3] = (unsigned)((ga >> 32) & 0x01ffffffu)            // global_addr[56:32]
            | (2u << 30);                                   // type = 2 ("image")
    const unsigned td0 = row_stride_elems;                  // tensor_dim0 (elements)
    const unsigned td1 = rows;                              // tensor_dim1
    s32x8 g1;
    g1[0] = (int)(1u << 16);                                // data_size=1 -> 2 bytes
    g1[1] = (int)((td0 & 0xffffu) << 16);                   // tensor_dim0[15:0]
    g1[2] = (int)(((td0 >> 16) & 0xffffu) | ((td1 & 0xffffu) << 16));
    g1[3] = (int)(((td1 >> 16) & 0xffffu) | ((tile_elems & 0xffffu) << 16)); // tile_dim0
    g1[4] = (int)(rows & 0xffffu);                          // tile_dim1 (tile_dim2=0)
    g1[5] = (int)row_stride_elems;                          // tensor_dim0_stride[31:0]
    g1[6] = 0;
    g1[7] = 0;
    const s32x4 z4 = {0, 0, 0, 0};
#if defined(__clang_major__) && __clang_major__ >= 23
    const s32x8 z8 = {0, 0, 0, 0, 0, 0, 0, 0};
    __builtin_amdgcn_tensor_load_to_lds(g0, g1, z4, z4, z8, 0);
#else
    __builtin_amdgcn_tensor_load_to_lds(g0, g1, z4, z4, 0);
#endif
}
#endif  // HAVE_TDM

// ---------------------------------------------------------------------------
// f32 -> bf16 conversion
// ---------------------------------------------------------------------------
__global__ __launch_bounds__(256)
void cvt_bf16_kernel(const float* __restrict__ in, __bf16* __restrict__ out, int n) {
    int i = blockIdx.x * 256 + threadIdx.x;
    if (i < n) out[i] = (__bf16)in[i];
}

// ---------------------------------------------------------------------------
// GEMM: C[m][n] = sum_k A[m][k] * W[n][k] + bias[n]
//   A: bf16 [M x K] row-major, W: bf16 [N x K] row-major.
//   Block = 256 threads = 8 waves, computes a 128(M) x 64(N) tile.
//   Wave w owns rows w*16..w*16+15 and all 4 column sub-tiles (A-frag reuse 4x,
//   B tile shared by all 8 waves via LDS).
//   TDM path: ping-pong double buffer; wave 0 keeps the next tile's two
//   tensor_load_to_lds in flight while all waves run WMMA on the current
//   buffer, waiting only for TENSORcnt <= 2 (TDM ops complete in order).
//   grid = (N/64, M/128).
// ---------------------------------------------------------------------------
template <bool OUT_F32>
__global__ __launch_bounds__(256)
void gemm_bf16_wmma(const __bf16* __restrict__ A, const __bf16* __restrict__ W,
                    const float* __restrict__ bias, void* __restrict__ Cout,
                    int M, int N, int K) {
    const int lane = threadIdx.x & 31;
    const int wave = threadIdx.x >> 5;
    const int l15  = lane & 15;
    const int half = lane >> 4;

    const int m0 = blockIdx.y * 128;
    const int n0 = blockIdx.x * 64;

    __shared__ __align__(16) __bf16 ldsA[2][128][32];   // 16 KB
    __shared__ __align__(16) __bf16 ldsB[2][64][32];    // 8 KB

    v8f acc[4];
#pragma unroll
    for (int j = 0; j < 4; ++j) acc[j] = (v8f){};

#if HAVE_TDM
    const unsigned ldsA_off[2] = { (unsigned)(size_t)(void*)&ldsA[0][0][0],
                                   (unsigned)(size_t)(void*)&ldsA[1][0][0] };
    const unsigned ldsB_off[2] = { (unsigned)(size_t)(void*)&ldsB[0][0][0],
                                   (unsigned)(size_t)(void*)&ldsB[1][0][0] };
    // Prologue: start DMA of the first k-tile into buffer 0.
    if (wave == 0) {
        tdm_load_tile_2d(A + (size_t)m0 * K, ldsA_off[0], 128, 32, (unsigned)K);
        tdm_load_tile_2d(W + (size_t)n0 * K, ldsB_off[0], 64,  32, (unsigned)K);
    }
    int buf = 0;
    for (int k0 = 0; k0 < K; k0 += 32) {
        if (wave == 0) {
            if (k0 + 32 < K) {
                // Issue next tile into the other buffer, then wait for the
                // current tile only (in-order completion => TENSORcnt <= 2).
                tdm_load_tile_2d(A + (size_t)m0 * K + k0 + 32, ldsA_off[buf ^ 1],
                                 128, 32, (unsigned)K);
                tdm_load_tile_2d(W + (size_t)n0 * K + k0 + 32, ldsB_off[buf ^ 1],
                                 64, 32, (unsigned)K);
                __builtin_amdgcn_s_wait_tensorcnt(2);
            } else {
                __builtin_amdgcn_s_wait_tensorcnt(0);
            }
        }
        __syncthreads();   // current buffer ready for all waves

        FragBF16 a;
        {
            const __bf16* la = &ldsA[buf][wave * 16 + l15][half * 8];
            a.q[0] = *(const uint4*)(la);
            a.q[1] = *(const uint4*)(la + 16);
        }
#pragma unroll
        for (int j = 0; j < 4; ++j) {
            const __bf16* lb = &ldsB[buf][j * 16 + l15][half * 8];
            FragBF16 b;
            b.q[0] = *(const uint4*)(lb);
            b.q[1] = *(const uint4*)(lb + 16);
            acc[j] = wmma_bf16(a, b, acc[j]);
        }

        __syncthreads();   // all reads of this buffer done before it's reused
        buf ^= 1;
    }
#else
    // Fallback: cooperative b128 copies, single buffer.
    for (int k0 = 0; k0 < K; k0 += 32) {
        __syncthreads();
        {
            int i = threadIdx.x;                    // 0..255 -> exactly A's 256 chunks
            int row = i >> 1, c = i & 1;
            *(uint4*)&ldsA[0][row][c * 16] =
                *(const uint4*)(A + (size_t)(m0 + row) * K + k0 + c * 16);
            if (i < 128) {
                row = i >> 1; c = i & 1;
                *(uint4*)&ldsB[0][row][c * 16] =
                    *(const uint4*)(W + (size_t)(n0 + row) * K + k0 + c * 16);
            }
        }
        __syncthreads();

        FragBF16 a;
        {
            const __bf16* la = &ldsA[0][wave * 16 + l15][half * 8];
            a.q[0] = *(const uint4*)(la);
            a.q[1] = *(const uint4*)(la + 16);
        }
#pragma unroll
        for (int j = 0; j < 4; ++j) {
            const __bf16* lb = &ldsB[0][j * 16 + l15][half * 8];
            FragBF16 b;
            b.q[0] = *(const uint4*)(lb);
            b.q[1] = *(const uint4*)(lb + 16);
            acc[j] = wmma_bf16(a, b, acc[j]);
        }
    }
#endif

#pragma unroll
    for (int j = 0; j < 4; ++j) {
        const int col = n0 + j * 16 + l15;
        const float bv = bias ? bias[col] : 0.0f;
#pragma unroll
        for (int r = 0; r < 8; ++r) {
            const int m = m0 + wave * 16 + r + 8 * half;   // C layout: row = r + 8*half
            const float v = acc[j][r] + bv;
            if (OUT_F32) ((float*)Cout)[(size_t)m * N + col] = v;
            else         ((__bf16*)Cout)[(size_t)m * N + col] = (__bf16)v;
        }
    }
}

// ---------------------------------------------------------------------------
// V transpose: Vt[b][h][d][l] = KV[b*L + l][D_MODEL + h*64 + d]   (bf16)
// ---------------------------------------------------------------------------
__global__ __launch_bounds__(256)
void transpose_v_kernel(const __bf16* __restrict__ KV, __bf16* __restrict__ Vt) {
    const unsigned total = B_BATCH * N_HEADS * D_HEAD * L_SEQ;  // 4,194,304
    unsigned idx = blockIdx.x * 256u + threadIdx.x;
    if (idx >= total) return;
    const int l = idx & (L_SEQ - 1);
    unsigned r = idx >> 12;
    const int d = r & (D_HEAD - 1); r >>= 6;
    const int h = r & (N_HEADS - 1);
    const int b = (int)(r >> 3);
    Vt[idx] = KV[((size_t)(b * L_SEQ + l)) * (2 * D_MODEL) + D_MODEL + h * D_HEAD + d];
}

// ---------------------------------------------------------------------------
// FlashAttention (forward, non-causal, additive mask).
//   Q  : bf16 [B*L][512]  (col offset h*64 selects head)
//   KV : bf16 [B*L][1024] (cols h*64..h*64+63 = K for head h)
//   Vt : bf16 [B*H][64][L]
//   mask: f32 [L][L]
//   Og : bf16 [B*L][512]  (attention output, pre final projection)
// grid = (B*H, L/128), block = 256; each wave owns 16 query rows.
// ---------------------------------------------------------------------------
__global__ __launch_bounds__(256)
void flash_attn_kernel(const __bf16* __restrict__ Q, const __bf16* __restrict__ KV,
                       const __bf16* __restrict__ Vt, const float* __restrict__ mask,
                       __bf16* __restrict__ Og) {
    const int lane = threadIdx.x & 31;
    const int wave = threadIdx.x >> 5;
    const int l15  = lane & 15;
    const int half = lane >> 4;

    const int bh = blockIdx.x;
    const int b  = bh >> 3;
    const int h  = bh & 7;
    const int q0 = (blockIdx.y * 8 + wave) * 16;

    // Q fragments (16 rows x 64 d) held in registers for the whole key loop.
    const __bf16* pq = Q + ((size_t)(b * L_SEQ + q0 + l15)) * D_MODEL + h * D_HEAD + half * 8;
    FragBF16 qa0, qa1;
    qa0.q[0] = *(const uint4*)(pq);        // d = half*8 .. +7
    qa0.q[1] = *(const uint4*)(pq + 16);   // d = 16 + half*8 .. +7
    qa1.q[0] = *(const uint4*)(pq + 32);   // d = 32 + ...
    qa1.q[1] = *(const uint4*)(pq + 48);   // d = 48 + ...

    v8f oacc[4];
#pragma unroll
    for (int t = 0; t < 4; ++t) oacc[t] = (v8f){};
    float Mx[8], Ls[8];
#pragma unroll
    for (int r = 0; r < 8; ++r) { Mx[r] = -1e30f; Ls[r] = 0.0f; }

    __shared__ __align__(16) __bf16 lds_p[8][16][32];   // per-wave P tile

    const float scale = 0.125f;   // 1/sqrt(64)
    const __bf16* vbase = Vt + (size_t)bh * D_HEAD * L_SEQ;

    for (int kt = 0; kt < L_SEQ; kt += 32) {
        const __bf16* pk = KV + ((size_t)(b * L_SEQ + kt + l15)) * (2 * D_MODEL)
                           + h * D_HEAD + half * 8;
        // Prefetch next key tile / V rows (global_prefetch_b8).
        if (kt + 32 < L_SEQ) {
            __builtin_prefetch(pk + (size_t)32 * (2 * D_MODEL), 0, 1);
            __builtin_prefetch(vbase + (size_t)l15 * L_SEQ + kt + 32, 0, 1);
        }

        // ---- S = Q * K^T for 32 keys (two 16-key column tiles) ----
        v8f s0 = (v8f){}, s1 = (v8f){};
        {
            FragBF16 kb0, kb1;
            kb0.q[0] = *(const uint4*)(pk);
            kb0.q[1] = *(const uint4*)(pk + 16);
            kb1.q[0] = *(const uint4*)(pk + 32);
            kb1.q[1] = *(const uint4*)(pk + 48);
            s0 = wmma_bf16(qa0, kb0, s0);
            s0 = wmma_bf16(qa1, kb1, s0);
            pk += (size_t)16 * (2 * D_MODEL);   // next 16 keys
            kb0.q[0] = *(const uint4*)(pk);
            kb0.q[1] = *(const uint4*)(pk + 16);
            kb1.q[0] = *(const uint4*)(pk + 32);
            kb1.q[1] = *(const uint4*)(pk + 48);
            s1 = wmma_bf16(qa0, kb0, s1);
            s1 = wmma_bf16(qa1, kb1, s1);
        }

        // ---- scale + mask, online softmax ----
#pragma unroll
        for (int r = 0; r < 8; ++r) {
            const int qm = q0 + r + 8 * half;
            s0[r] = s0[r] * scale + mask[(size_t)qm * L_SEQ + kt + l15];
            s1[r] = s1[r] * scale + mask[(size_t)qm * L_SEQ + kt + 16 + l15];
        }
#pragma unroll
        for (int r = 0; r < 8; ++r) {
            float tmax = halfgrp_max(fmaxf(s0[r], s1[r]));
            const float nm = fmaxf(Mx[r], tmax);
            const float alpha = __expf(Mx[r] - nm);
            Mx[r] = nm;
            const float p0 = __expf(s0[r] - nm);
            const float p1 = __expf(s1[r] - nm);
            s0[r] = p0;
            s1[r] = p1;
            const float rs = halfgrp_sum(p0 + p1);
            Ls[r] = Ls[r] * alpha + rs;
            oacc[0][r] *= alpha;
            oacc[1][r] *= alpha;
            oacc[2][r] *= alpha;
            oacc[3][r] *= alpha;
        }

        // ---- P: C-fragment layout -> A-fragment layout via LDS ----
#pragma unroll
        for (int r = 0; r < 8; ++r) {
            const int m = r + 8 * half;
            lds_p[wave][m][l15]      = (__bf16)s0[r];
            lds_p[wave][m][16 + l15] = (__bf16)s1[r];
        }
        asm volatile("s_wait_dscnt 0" ::: "memory");
        FragBF16 pa;
        {
            const __bf16* lp = &lds_p[wave][l15][half * 8];
            pa.q[0] = *(const uint4*)(lp);
            pa.q[1] = *(const uint4*)(lp + 16);
        }

        // ---- O += P * V  (4 d-tiles of 16) ----
#pragma unroll
        for (int dt = 0; dt < 4; ++dt) {
            const __bf16* pv = vbase + ((size_t)(dt * 16 + l15)) * L_SEQ + kt + half * 8;
            FragBF16 vb;
            vb.q[0] = *(const uint4*)(pv);
            vb.q[1] = *(const uint4*)(pv + 16);
            oacc[dt] = wmma_bf16(pa, vb, oacc[dt]);
        }
    }

    // ---- normalize rows and store bf16 attention output ----
#pragma unroll
    for (int r = 0; r < 8; ++r) {
        const float inv = 1.0f / Ls[r];
        const int m = b * L_SEQ + q0 + r + 8 * half;
        __bf16* po = Og + (size_t)m * D_MODEL + h * D_HEAD;
        po[0 * 16 + l15] = (__bf16)(oacc[0][r] * inv);
        po[1 * 16 + l15] = (__bf16)(oacc[1][r] * inv);
        po[2 * 16 + l15] = (__bf16)(oacc[2][r] * inv);
        po[3 * 16 + l15] = (__bf16)(oacc[3][r] * inv);
    }
}

// ---------------------------------------------------------------------------
// Host-side launch
// ---------------------------------------------------------------------------
extern "C" void kernel_launch(void* const* d_in, const int* in_sizes, int n_in,
                              void* d_out, int out_size, void* d_ws, size_t ws_size,
                              hipStream_t stream) {
    (void)in_sizes; (void)n_in; (void)out_size; (void)ws_size;

    const float* x    = (const float*)d_in[0];
    const float* mask = (const float*)d_in[1];
    const float* q_w  = (const float*)d_in[2];
    const float* q_b  = (const float*)d_in[3];
    const float* kv_w = (const float*)d_in[4];
    const float* kv_b = (const float*)d_in[5];
    const float* f_w  = (const float*)d_in[6];
    const float* f_b  = (const float*)d_in[7];
    float* out = (float*)d_out;

    const int M = B_BATCH * L_SEQ;          // 8192
    const int D = D_MODEL;                  // 512

    // Workspace carve-up (256B aligned).
    char* ws = (char*)d_ws;
    size_t off = 0;
    auto carve = [&](size_t bytes) { void* p = ws + off; off += (bytes + 255) & ~(size_t)255; return p; };
    __bf16* xb   = (__bf16*)carve((size_t)M * D * 2);            // x bf16
    __bf16* qwb  = (__bf16*)carve((size_t)D * D * 2);            // q_w bf16
    __bf16* kvwb = (__bf16*)carve((size_t)2 * D * D * 2);        // kv_w bf16
    __bf16* fwb  = (__bf16*)carve((size_t)D * D * 2);            // f_w bf16
    __bf16* Qt   = (__bf16*)carve((size_t)M * D * 2);            // Q bf16 [M][512]
    __bf16* KVt  = (__bf16*)carve((size_t)M * 2 * D * 2);        // KV bf16 [M][1024]
    __bf16* Vt   = (__bf16*)carve((size_t)M * D * 2);            // V^T bf16 [B*H][64][L]
    __bf16* AO   = (__bf16*)carve((size_t)M * D * 2);            // attn out bf16 [M][512]

    // 1) f32 -> bf16 conversions
    {
        int n;
        n = M * D;         cvt_bf16_kernel<<<(n + 255) / 256, 256, 0, stream>>>(x,    xb,   n);
        n = D * D;         cvt_bf16_kernel<<<(n + 255) / 256, 256, 0, stream>>>(q_w,  qwb,  n);
        n = 2 * D * D;     cvt_bf16_kernel<<<(n + 255) / 256, 256, 0, stream>>>(kv_w, kvwb, n);
        n = D * D;         cvt_bf16_kernel<<<(n + 255) / 256, 256, 0, stream>>>(f_w,  fwb,  n);
    }

    // 2) projections: Q = x q_w^T + q_b ; KV = x kv_w^T + kv_b
    gemm_bf16_wmma<false><<<dim3(D / 64, M / 128), 256, 0, stream>>>(xb, qwb,  q_b,  Qt,  M, D,     D);
    gemm_bf16_wmma<false><<<dim3(2 * D / 64, M / 128), 256, 0, stream>>>(xb, kvwb, kv_b, KVt, M, 2 * D, D);

    // 3) V transpose into [B*H][64][L]
    {
        const unsigned total = (unsigned)B_BATCH * N_HEADS * D_HEAD * L_SEQ;
        transpose_v_kernel<<<(total + 255) / 256, 256, 0, stream>>>(KVt, Vt);
    }

    // 4) attention
    flash_attn_kernel<<<dim3(B_BATCH * N_HEADS, L_SEQ / 128), 256, 0, stream>>>(Qt, KVt, Vt, mask, AO);

    // 5) final projection: out = AO f_w^T + f_b  (f32 output)
    gemm_bf16_wmma<true><<<dim3(D / 64, M / 128), 256, 0, stream>>>(AO, fwb, f_b, out, M, D, D);
}
